// Longformer_41669772705909
// MI455X (gfx1250) — matrix operand order repaired and checked by reading, not compiled, not run on hardware
//
#include <hip/hip_runtime.h>
#include <cstdint>

#define NH_    12
#define DH_    64
#define DM_    768
#define FD_    3072
#define SEQ_   4096
#define BATCH_ 2
#define WIN_   128
#define NG_    32
#define NEGV   (-1.0e9f)

typedef __attribute__((ext_vector_type(16))) __bf16 bf16x16;
typedef __attribute__((ext_vector_type(8)))  float  floatx8;
typedef __attribute__((ext_vector_type(4)))  int    i32x4;

union FragU { bf16x16 v; unsigned u[8]; };

__device__ __forceinline__ unsigned short f2bf(float x) {
  union { float f; unsigned u; } c; c.f = x;
  unsigned u = c.u;
  u += 0x7fffu + ((u >> 16) & 1u);          // round-to-nearest-even
  return (unsigned short)(u >> 16);
}
__device__ __forceinline__ float bf2f(unsigned short h) {
  union { float f; unsigned u; } c; c.u = ((unsigned)h) << 16;
  return c.f;
}
__device__ __forceinline__ floatx8 zero8() {
  floatx8 z;
#pragma unroll
  for (int i = 0; i < 8; ++i) z[i] = 0.0f;
  return z;
}
__device__ __forceinline__ floatx8 wmma_bf16(bf16x16 a, bf16x16 b, floatx8 c) {
  return __builtin_amdgcn_wmma_f32_16x16x32_bf16(false, a, false, b, (short)0, c,
                                                 false, false);
}
// A fragment: lane&15 = row m; lanes>=16 take K+16..31; VGPR j = K pair 2j,2j+1.
__device__ __forceinline__ bf16x16 frag_a(const unsigned short* sm, int stride,
                                          int mbase, int kbase) {
  const int lane = threadIdx.x & 31;
  const int grp  = lane >> 4;
  const unsigned short* row = sm + (mbase + (lane & 15)) * stride;
  FragU f;
#pragma unroll
  for (int j = 0; j < 8; ++j) {
    int kk = kbase + grp * 16 + 2 * j;
    f.u[j] = *(const unsigned*)(row + kk);
  }
  return f.v;
}
// B fragment: lane&15 = col n; lanes>=16 take K+16..31; VGPR j = K pair 2j,2j+1.
__device__ __forceinline__ bf16x16 frag_b(const unsigned short* sm, int stride,
                                          int nbase, int kbase) {
  const int lane = threadIdx.x & 31;
  const int grp  = lane >> 4;
  const unsigned short* row = sm + (nbase + (lane & 15)) * stride;
  FragU f;
#pragma unroll
  for (int j = 0; j < 8; ++j) {
    int kk = kbase + grp * 16 + 2 * j;
    f.u[j] = *(const unsigned*)(row + kk);
  }
  return f.v;
}
__device__ __forceinline__ float wave_max(float v) {
#pragma unroll
  for (int m = 16; m > 0; m >>= 1) v = fmaxf(v, __shfl_xor(v, m, 32));
  return v;
}
__device__ __forceinline__ float wave_sum(float v) {
#pragma unroll
  for (int m = 16; m > 0; m >>= 1) v += __shfl_xor(v, m, 32);
  return v;
}

// ---- CDNA5 async global->LDS copy (bypasses VGPRs, ASYNCcnt-tracked) -------
#if __has_builtin(__builtin_amdgcn_global_load_async_to_lds_b128) && \
    __has_builtin(__builtin_amdgcn_s_wait_asynccnt)
__device__ __forceinline__ void copy16_g2l(const void* g, void* l) {
  __builtin_amdgcn_global_load_async_to_lds_b128((i32x4*)g, (i32x4*)l, 0, 0);
}
__device__ __forceinline__ void copy_wait() { __builtin_amdgcn_s_wait_asynccnt(0); }
#else
__device__ __forceinline__ void copy16_g2l(const void* g, void* l) {
  *(uint4*)l = *(const uint4*)g;   // call sites are 16B-aligned both sides
}
__device__ __forceinline__ void copy_wait() {}
#endif

// ---------------------------------------------------------------------------
// WMMA GEMM: C[M,N] = A[M,K](f32) @ W[K,N](f32), bf16 compute, fp32 accum.
// 128x128 block tile, K-step 32, 8 waves = 4(M)x2(N), 2x4 wmma tiles/wave.
// Software-pipelined staging + global_prefetch of tile k+2.
// EPI 0: bf16 head-major [B,NH,S,64], (acc+bias)*scale   (QKV)
// EPI 1: f32 = acc + bias + resid                        (pre-LN)
// EPI 2: f32 = gelu_exact(acc + bias)                    (FFN1)
// ---------------------------------------------------------------------------
#define GSTR 34   // 32 + 2 pad -> 17-dword row stride

template <int EPI>
__global__ __launch_bounds__(256) void gemm_wmma(
    const float* __restrict__ A, const float* __restrict__ Bw,
    const float* __restrict__ bias, const float* __restrict__ resid,
    void* __restrict__ outp, int M, int N, int K, float scale) {
  __shared__ unsigned short As[128 * GSTR];
  __shared__ unsigned short Bs[128 * GSTR];
  const int tid  = threadIdx.x;
  const int wave = tid >> 5;
  const int lane = tid & 31;
  const int grp  = lane >> 4, nl = lane & 15;
  const int m0 = blockIdx.y * 128;
  const int n0 = blockIdx.x * 128;
  const int waveM = wave & 3;
  const int waveN = wave >> 2;
  const int ar = tid >> 3, ac = (tid & 7) * 4;   // A staging coords
  const int bk = tid >> 5, bn = (tid & 31) * 4;  // B staging coords
  floatx8 acc[2][4];
#pragma unroll
  for (int i = 0; i < 2; ++i)
#pragma unroll
    for (int j = 0; j < 4; ++j) acc[i][j] = zero8();

  float4 rA[4], rB[4];
  auto load_tiles = [&](int kt) {
#pragma unroll
    for (int p = 0; p < 4; ++p)
      rA[p] = *(const float4*)(A + (size_t)(m0 + ar + 32 * p) * K + kt + ac);
#pragma unroll
    for (int p = 0; p < 4; ++p)
      rB[p] = *(const float4*)(Bw + (size_t)(kt + bk + 8 * p) * N + n0 + bn);
  };
  load_tiles(0);

  for (int kt = 0; kt < K; kt += 32) {
#pragma unroll
    for (int p = 0; p < 4; ++p) {
      unsigned* dst = (unsigned*)(As + (ar + 32 * p) * GSTR + ac);
      dst[0] = (unsigned)f2bf(rA[p].x) | ((unsigned)f2bf(rA[p].y) << 16);
      dst[1] = (unsigned)f2bf(rA[p].z) | ((unsigned)f2bf(rA[p].w) << 16);
    }
#pragma unroll
    for (int p = 0; p < 4; ++p) {
      int kk = bk + 8 * p;
      Bs[(bn + 0) * GSTR + kk] = f2bf(rB[p].x);
      Bs[(bn + 1) * GSTR + kk] = f2bf(rB[p].y);
      Bs[(bn + 2) * GSTR + kk] = f2bf(rB[p].z);
      Bs[(bn + 3) * GSTR + kk] = f2bf(rB[p].w);
    }
    __syncthreads();
    if (kt + 32 < K) load_tiles(kt + 32);             // overlap with WMMA below
    if (kt + 64 < K) {                                 // gfx1250 global_prefetch
      __builtin_prefetch(A + (size_t)(m0 + ar) * K + kt + 64 + ac, 0, 1);
      __builtin_prefetch(Bw + (size_t)(kt + 64 + bk) * N + n0 + bn, 0, 1);
    }
    bf16x16 af[2], bfr[4];
#pragma unroll
    for (int mt = 0; mt < 2; ++mt) af[mt] = frag_a(As, GSTR, waveM * 32 + mt * 16, 0);
#pragma unroll
    for (int nt = 0; nt < 4; ++nt) bfr[nt] = frag_b(Bs, GSTR, waveN * 64 + nt * 16, 0);
#pragma unroll
    for (int mt = 0; mt < 2; ++mt)
#pragma unroll
      for (int nt = 0; nt < 4; ++nt) acc[mt][nt] = wmma_bf16(af[mt], bfr[nt], acc[mt][nt]);
    __syncthreads();
  }
#pragma unroll
  for (int mt = 0; mt < 2; ++mt)
#pragma unroll
    for (int nt = 0; nt < 4; ++nt)
#pragma unroll
      for (int i = 0; i < 8; ++i) {
        int row = m0 + waveM * 32 + mt * 16 + grp * 8 + i;
        int col = n0 + waveN * 64 + nt * 16 + nl;
        float v = acc[mt][nt][i];
        if constexpr (EPI == 0) {
          v = (v + bias[col]) * scale;
          int hh = col >> 6, dd = col & 63;
          int bb = row / SEQ_, ss = row - bb * SEQ_;
          ((unsigned short*)outp)[(((size_t)(bb * NH_ + hh) * SEQ_ + ss) << 6) + dd] = f2bf(v);
        } else if constexpr (EPI == 1) {
          ((float*)outp)[(size_t)row * N + col] = v + bias[col] + resid[(size_t)row * N + col];
        } else {
          v += bias[col];
          ((float*)outp)[(size_t)row * N + col] = 0.5f * v * (1.0f + erff(v * 0.70710678118f));
        }
      }
}

// ---------------------------------------------------------------------------
// Banded attention: block per (b, h, chunk, 64-q-row half).
// Q/K staged via async b128 copies (16B-aligned rows, stride 72);
// am pre-staged into LDS so the score epilogue has no global traffic.
// ---------------------------------------------------------------------------
#define QKSTR 72   // row stride for Qs/Ks: 144 B (16B aligned), 36 dwords

__global__ __launch_bounds__(256) void band_attn(
    const unsigned short* __restrict__ qb, const unsigned short* __restrict__ kb,
    const unsigned short* __restrict__ vb, const float* __restrict__ am,
    float* __restrict__ ctx) {
  extern __shared__ unsigned short bsm[];
  unsigned short* Qs = bsm;                  // 64  x 72
  unsigned short* Ks = Qs + 64 * QKSTR;      // 416 x 72  (n-major, Dh contig)
  unsigned short* Ps = Ks + 416 * QKSTR;     // 64  x 418 (scores -> probs)
  unsigned short* Vs = Ps + 64 * 418;        // 64  x 418 (V transposed [d][n])
  float*          Am = (float*)(Vs + 64 * 418); // 416 mask adds
  const int tid = threadIdx.x, wave = tid >> 5, lane = tid & 31;
  const int grp = lane >> 4, nl = lane & 15;
  int bid = blockIdx.x;
  const int half = bid & 1;  bid >>= 1;
  const int c    = bid & 31; bid >>= 5;
  const int h    = bid % NH_;
  const int b    = bid / NH_;
  const int qoff = half * 64;
  const size_t headO = ((size_t)(b * NH_ + h)) * SEQ_ * DH_;
  const unsigned short* qh = qb + headO;
  const unsigned short* kh = kb + headO;
  const unsigned short* vh = vb + headO;

  for (int idx = tid; idx < 64 * 8; idx += 256) {          // Q: 8 x b128 / row
    int r = idx >> 3, seg = (idx & 7) * 8;
    int s = c * WIN_ + qoff + r;
    copy16_g2l(qh + (size_t)s * DH_ + seg, Qs + r * QKSTR + seg);
  }
  for (int idx = tid; idx < 416 * 8; idx += 256) {         // K band+global
    int n = idx >> 3, seg = (idx & 7) * 8;
    int s; bool ok;
    if (n < 384) { s = c * WIN_ - WIN_ + n; ok = (s >= 0) && (s < SEQ_); }
    else         { s = n - 384;             ok = true; }
    if (ok) copy16_g2l(kh + (size_t)s * DH_ + seg, Ks + n * QKSTR + seg);
    else    *(uint4*)(Ks + n * QKSTR + seg) = uint4{0u, 0u, 0u, 0u};
  }
  for (int idx = tid; idx < 416 * 64; idx += 256) {        // V transposed
    int n = idx >> 6, d = idx & 63;
    int s; bool ok;
    if (n < 384) { s = c * WIN_ - WIN_ + n; ok = (s >= 0) && (s < SEQ_); }
    else         { s = n - 384;             ok = true; }
    Vs[d * 418 + n] = ok ? vh[(size_t)s * DH_ + d] : (unsigned short)0;
  }
  for (int n = tid; n < 416; n += 256) {                   // additive mask
    float a = 0.f;
    if (n < 384) {
      int kpos = c * WIN_ - WIN_ + n;
      if (kpos >= 0 && kpos < SEQ_) a = am[(size_t)b * SEQ_ + kpos];
    } else {
      a = am[(size_t)b * SEQ_ + (n - 384)];
    }
    Am[n] = a;
  }
  copy_wait();
  __syncthreads();
  // scores: 4 m-tiles x 26 n-tiles = 104 wmma tiles over 8 waves
  for (int tile = wave; tile < 104; tile += 8) {
    int mt = tile & 3, nt = tile >> 2;
    floatx8 acc = zero8();
#pragma unroll
    for (int ks = 0; ks < 2; ++ks)
      acc = wmma_bf16(frag_a(Qs, QKSTR, mt * 16, ks * 32),
                      frag_b(Ks, QKSTR, nt * 16, ks * 32), acc);
    int n = nt * 16 + nl;
    float amv = Am[n];
#pragma unroll
    for (int i = 0; i < 8; ++i) {
      int m = mt * 16 + grp * 8 + i;
      int qrow = qoff + m;
      float val = acc[i] + amv;
      if (n < 384) {
        int kpos = c * WIN_ - WIN_ + n;
        bool valid = ((unsigned)(n - qrow) <= 256u) && (kpos >= NG_) && (kpos < SEQ_);
        val = valid ? val : NEGV;
      }
      Ps[m * 418 + n] = f2bf(val);
    }
  }
  __syncthreads();
  // softmax over 416 cols, 8 rows per wave
  for (int rr = 0; rr < 8; ++rr) {
    int r = wave * 8 + rr;
    float vals[13];
#pragma unroll
    for (int j = 0; j < 13; ++j) vals[j] = bf2f(Ps[r * 418 + lane + 32 * j]);
    float mx = vals[0];
#pragma unroll
    for (int j = 1; j < 13; ++j) mx = fmaxf(mx, vals[j]);
    mx = wave_max(mx);
    float sum = 0.f;
#pragma unroll
    for (int j = 0; j < 13; ++j) { vals[j] = __expf(vals[j] - mx); sum += vals[j]; }
    sum = wave_sum(sum);
    float inv = 1.0f / sum;
#pragma unroll
    for (int j = 0; j < 13; ++j) Ps[r * 418 + lane + 32 * j] = f2bf(vals[j] * inv);
  }
  __syncthreads();
  // out = P(64x416) @ V(416x64): 16 tiles, K = 13 x 32
  for (int tile = wave; tile < 16; tile += 8) {
    int mt = tile & 3, nt = tile >> 2;
    floatx8 acc = zero8();
#pragma unroll
    for (int ks = 0; ks < 13; ++ks)
      acc = wmma_bf16(frag_a(Ps, 418, mt * 16, ks * 32),
                      frag_b(Vs, 418, nt * 16, ks * 32), acc);
#pragma unroll
    for (int i = 0; i < 8; ++i) {
      int s   = c * WIN_ + qoff + mt * 16 + grp * 8 + i;
      int col = h * DH_ + nt * 16 + nl;
      ctx[((size_t)b * SEQ_ + s) * DM_ + col] = acc[i];
    }
  }
}

// ---------------------------------------------------------------------------
// Global-query attention: 32 q rows vs full S, flash-style, 256-key chunks.
// One block per (b,h). Overwrites ctx rows s < 32.
// ---------------------------------------------------------------------------
__global__ __launch_bounds__(256) void globalq_attn(
    const unsigned short* __restrict__ qb, const unsigned short* __restrict__ kb,
    const unsigned short* __restrict__ vb, const float* __restrict__ am,
    float* __restrict__ ctx) {
  extern __shared__ char gsmc[];
  float* Sf   = (float*)gsmc;                // 32 x 260 chunk scores
  float* Oa   = Sf + 32 * 260;               // 32 x 64 output accum
  float* mrow = Oa + 32 * 64;                // 32
  float* lrow = mrow + 32;                   // 32
  float* Am   = lrow + 32;                   // 256 chunk mask adds
  unsigned short* Qs = (unsigned short*)(Am + 256); // 32 x 72
  unsigned short* Ks = Qs + 32 * QKSTR;      // 256 x 72
  unsigned short* Vs = Ks + 256 * QKSTR;     // 64 x 258 (transposed)
  unsigned short* Pb = Vs + 64 * 258;        // 32 x 258 probs bf16
  const int tid = threadIdx.x, wave = tid >> 5, lane = tid & 31;
  const int grp = lane >> 4, nl = lane & 15;
  const int h = blockIdx.x % NH_, b = blockIdx.x / NH_;
  const size_t headO = ((size_t)(b * NH_ + h)) * SEQ_ * DH_;
  const unsigned short* qh = qb + headO;
  const unsigned short* kh = kb + headO;
  const unsigned short* vh = vb + headO;

  {                                           // Q: 32 rows x 8 b128 segs
    int r = tid >> 3, seg = (tid & 7) * 8;
    copy16_g2l(qh + (size_t)r * DH_ + seg, Qs + r * QKSTR + seg);
  }
  for (int idx = tid; idx < 32 * 64; idx += 256) Oa[idx] = 0.f;
  if (tid < 32) { mrow[tid] = -3.0e38f; lrow[tid] = 0.f; }

  for (int ck = 0; ck < SEQ_ / 256; ++ck) {
    const int s0 = ck * 256;
    __syncthreads();
    for (int idx = tid; idx < 256 * 8; idx += 256) {
      int n = idx >> 3, seg = (idx & 7) * 8;
      copy16_g2l(kh + (size_t)(s0 + n) * DH_ + seg, Ks + n * QKSTR + seg);
    }
    for (int idx = tid; idx < 256 * 64; idx += 256) {
      int n = idx >> 6, d = idx & 63;
      Vs[d * 258 + n] = vh[(size_t)(s0 + n) * DH_ + d];
    }
    Am[tid] = am[(size_t)b * SEQ_ + s0 + tid];
    copy_wait();
    __syncthreads();
    for (int tile = wave; tile < 32; tile += 8) {   // scores: 2m x 16n tiles
      int mt = tile & 1, nt = tile >> 1;
      floatx8 acc = zero8();
#pragma unroll
      for (int ks = 0; ks < 2; ++ks)
        acc = wmma_bf16(frag_a(Qs, QKSTR, mt * 16, ks * 32),
                        frag_b(Ks, QKSTR, nt * 16, ks * 32), acc);
      int n = nt * 16 + nl;
      float amv = Am[n];
#pragma unroll
      for (int i = 0; i < 8; ++i)
        Sf[(mt * 16 + grp * 8 + i) * 260 + n] = acc[i] + amv;
    }
    __syncthreads();
    for (int rr = 0; rr < 4; ++rr) {                // online softmax, 4 rows/wave
      int r = wave * 4 + rr;
      float vals[8];
#pragma unroll
      for (int j = 0; j < 8; ++j) vals[j] = Sf[r * 260 + lane + 32 * j];
      float cmax = vals[0];
#pragma unroll
      for (int j = 1; j < 8; ++j) cmax = fmaxf(cmax, vals[j]);
      cmax = wave_max(cmax);
      float mold = mrow[r];
      float mnew = fmaxf(mold, cmax);
      float alpha = __expf(mold - mnew);
      float csum = 0.f;
#pragma unroll
      for (int j = 0; j < 8; ++j) { vals[j] = __expf(vals[j] - mnew); csum += vals[j]; }
      csum = wave_sum(csum);
      if (lane == 0) { lrow[r] = lrow[r] * alpha + csum; mrow[r] = mnew; }
#pragma unroll
      for (int j = 0; j < 8; ++j) Pb[r * 258 + lane + 32 * j] = f2bf(vals[j]);
      Oa[r * 64 + lane]      *= alpha;
      Oa[r * 64 + 32 + lane] *= alpha;
    }
    __syncthreads();
    {                                               // O += P @ V (8 tiles)
      int mt = wave & 1, nt = wave >> 1;
      floatx8 acc = zero8();
#pragma unroll
      for (int ks = 0; ks < 8; ++ks)
        acc = wmma_bf16(frag_a(Pb, 258, mt * 16, ks * 32),
                        frag_b(Vs, 258, nt * 16, ks * 32), acc);
#pragma unroll
      for (int i = 0; i < 8; ++i)
        Oa[(mt * 16 + grp * 8 + i) * 64 + nt * 16 + nl] += acc[i];
    }
  }
  __syncthreads();
  for (int idx = tid; idx < 32 * 64; idx += 256) {
    int r = idx >> 6, d = idx & 63;
    ctx[((size_t)b * SEQ_ + r) * DM_ + h * DH_ + d] = Oa[idx] / lrow[r];
  }
}

// ---------------------------------------------------------------------------
__global__ __launch_bounds__(256) void layernorm_k(
    const float* __restrict__ x, const float* __restrict__ g,
    const float* __restrict__ bta, float* __restrict__ out, int N) {
  __shared__ float red[256];
  const int tid = threadIdx.x;
  const float* xr = x + (size_t)blockIdx.x * N;
  float s = 0.f;
  for (int i = tid; i < N; i += 256) s += xr[i];
  red[tid] = s; __syncthreads();
  for (int st = 128; st > 0; st >>= 1) { if (tid < st) red[tid] += red[tid + st]; __syncthreads(); }
  const float mu = red[0] / (float)N;
  __syncthreads();
  float vs = 0.f;
  for (int i = tid; i < N; i += 256) { float d = xr[i] - mu; vs += d * d; }
  red[tid] = vs; __syncthreads();
  for (int st = 128; st > 0; st >>= 1) { if (tid < st) red[tid] += red[tid + st]; __syncthreads(); }
  const float inv = rsqrtf(red[0] / (float)N + 1e-12f);
  float* orow = out + (size_t)blockIdx.x * N;
  for (int i = tid; i < N; i += 256) orow[i] = (xr[i] - mu) * inv * g[i] + bta[i];
}

// ---------------------------------------------------------------------------
extern "C" void kernel_launch(void* const* d_in, const int* in_sizes, int n_in,
                              void* d_out, int out_size, void* d_ws, size_t ws_size,
                              hipStream_t stream) {
  const float* hx  = (const float*)d_in[0];
  const float* am  = (const float*)d_in[1];
  const float* Wq  = (const float*)d_in[2];
  const float* bq  = (const float*)d_in[3];
  const float* Wk  = (const float*)d_in[4];
  const float* bk  = (const float*)d_in[5];
  const float* Wv  = (const float*)d_in[6];
  const float* bv  = (const float*)d_in[7];
  const float* Wo  = (const float*)d_in[8];
  const float* bo  = (const float*)d_in[9];
  const float* g1  = (const float*)d_in[10];
  const float* b1  = (const float*)d_in[11];
  const float* Wi  = (const float*)d_in[12];
  const float* bi  = (const float*)d_in[13];
  const float* Wo2 = (const float*)d_in[14];
  const float* bo2 = (const float*)d_in[15];
  const float* g2  = (const float*)d_in[16];
  const float* b2  = (const float*)d_in[17];

  const int M = BATCH_ * SEQ_;  // 8192
  char* ws = (char*)d_ws;
  size_t off = 0;
  auto salloc = [&](size_t bytes) {
    void* p = ws + off;
    off = (off + bytes + 255) & ~(size_t)255;
    return p;
  };
  unsigned short* qbuf = (unsigned short*)salloc((size_t)BATCH_ * NH_ * SEQ_ * DH_ * 2);
  unsigned short* kbuf = (unsigned short*)salloc((size_t)BATCH_ * NH_ * SEQ_ * DH_ * 2);
  unsigned short* vbuf = (unsigned short*)salloc((size_t)BATCH_ * NH_ * SEQ_ * DH_ * 2);
  float* ctx   = (float*)salloc((size_t)M * DM_ * 4);
  float* pre1  = (float*)salloc((size_t)M * DM_ * 4);
  float* aout  = (float*)salloc((size_t)M * DM_ * 4);
  float* inter = (float*)salloc((size_t)M * FD_ * 4);
  float* pre2  = ctx;  // ctx dead after Wo GEMM -> reuse

  const size_t BAND_LDS = (size_t)(64 * QKSTR + 416 * QKSTR + 2 * 64 * 418) * 2 +
                          (size_t)416 * 4;                        // 177792 B
  const size_t GQ_LDS   = (size_t)4 * (32 * 260 + 32 * 64 + 64 + 256) +
                          (size_t)2 * (32 * QKSTR + 256 * QKSTR + 64 * 258 + 32 * 258); // 133760 B

  dim3 blk(256, 1, 1);
  dim3 gD(DM_ / 128, M / 128, 1);
  dim3 gF(FD_ / 128, M / 128, 1);

  gemm_wmma<0><<<gD, blk, 0, stream>>>(hx, Wq, bq, nullptr, qbuf, M, DM_, DM_, 0.125f);
  gemm_wmma<0><<<gD, blk, 0, stream>>>(hx, Wk, bk, nullptr, kbuf, M, DM_, DM_, 1.0f);
  gemm_wmma<0><<<gD, blk, 0, stream>>>(hx, Wv, bv, nullptr, vbuf, M, DM_, DM_, 1.0f);

  band_attn<<<dim3(BATCH_ * NH_ * 32 * 2, 1, 1), blk, BAND_LDS, stream>>>(qbuf, kbuf, vbuf, am, ctx);
  globalq_attn<<<dim3(BATCH_ * NH_, 1, 1), blk, GQ_LDS, stream>>>(qbuf, kbuf, vbuf, am, ctx);

  gemm_wmma<1><<<gD, blk, 0, stream>>>(ctx, Wo, bo, hx, pre1, M, DM_, DM_, 1.0f);
  layernorm_k<<<dim3(M, 1, 1), blk, 0, stream>>>(pre1, g1, b1, aout, DM_);
  gemm_wmma<2><<<gF, blk, 0, stream>>>(aout, Wi, bi, nullptr, inter, M, FD_, DM_, 1.0f);
  gemm_wmma<1><<<gD, blk, 0, stream>>>(inter, Wo2, bo2, aout, pre2, M, DM_, FD_, 1.0f);
  layernorm_k<<<dim3(M, 1, 1), blk, 0, stream>>>(pre2, g2, b2, (float*)d_out, DM_);
}
